// ExplicitGeometricAugmentor_78271484002640
// MI455X (gfx1250) — matrix operand order repaired and checked by reading, not compiled, not run on hardware
//
#include <hip/hip_runtime.h>
#include <hip/hip_bf16.h>

typedef __attribute__((ext_vector_type(16))) __bf16 bf16x16;
typedef __attribute__((ext_vector_type(8)))  float  v8f;

#define B_SZ   32
#define NPAD   208
#define NPATCH 196
#define MTOT   (B_SZ * NPAD)   // 6656
#define DIMD   768
#define MLPD   3072

__device__ __forceinline__ unsigned short f2bf(float f) {
  unsigned u = __builtin_bit_cast(unsigned, f);
  unsigned r = u + 0x7FFFu + ((u >> 16) & 1u);
  return (unsigned short)(r >> 16);
}

__device__ __forceinline__ unsigned pack2bf(float lo, float hi) {
  return (unsigned)f2bf(lo) | ((unsigned)f2bf(hi) << 16);
}

__device__ __forceinline__ float softplus_f(float x) {
  return fmaxf(x, 0.0f) + log1pf(expf(-fabsf(x)));
}

// ---------------------------------------------------------------------------
// Generic batched GEMM: C[b] = act( A[b](bf16, MxK) * B[b](f32->bf16, KxN) + bias )
// Block tile 128x64, K-step 32, 8 waves; each wave owns a 32x32 register tile
// -> 4 x v_wmma_f32_16x16x32_bf16 per wave per K-step (A/B fragment reuse).
// ---------------------------------------------------------------------------
__global__ __launch_bounds__(256)
void gemm_bf16_wmma(const unsigned short* __restrict__ A, int lda, long long sA,
                    const float* __restrict__ Bm, int ldb, long long sB,
                    const float* __restrict__ bias,
                    const float* __restrict__ resid,
                    float* __restrict__ out32,
                    unsigned short* __restrict__ out16,
                    long long sC, int M, int N, int K, int act_gelu)
{
  __shared__ __align__(16) unsigned short lsA[128 * 40];
  __shared__ __align__(16) unsigned short lsB[64 * 40];

  const int tid  = threadIdx.x;
  const int bz   = blockIdx.z;
  const int row0 = blockIdx.y * 128;
  const int col0 = blockIdx.x * 64;

  const unsigned short* Ab = A  + (long long)bz * sA;
  const float*          Bb = Bm + (long long)bz * sB;

  const int lane = tid & 31;
  const int wave = tid >> 5;
  const int wm   = wave >> 1;   // 0..3 : 32-row slice
  const int wn   = wave & 1;    // 0..1 : 32-col slice
  const int lm   = lane & 15;
  const int lh   = lane >> 4;

  v8f acc[2][2];
  #pragma unroll
  for (int i = 0; i < 2; ++i)
    #pragma unroll
    for (int j = 0; j < 2; ++j)
      acc[i][j] = (v8f){0.f,0.f,0.f,0.f,0.f,0.f,0.f,0.f};

  // A staging: thread -> row (tid>>1) in [0,128), k-chunk (tid&1)*16
  const int ar  = tid >> 1;
  const int akc = (tid & 1) << 4;        // 0 or 16
  // B staging: thread -> k-pair (tid&15)*2, n-quad (tid>>4)*4
  const int bkp = (tid & 15) << 1;       // even k in [0,32)
  const int bng = (tid >> 4) << 2;       // n in [0,64) step 4

  const int ksteps = (K + 31) >> 5;
  for (int kt = 0; kt < ksteps; ++kt) {
    const int k0 = kt << 5;

    // ---- stage A tile (bf16, 2 x 16B per thread) ----
    {
      uint4 q0 = make_uint4(0u,0u,0u,0u), q1 = make_uint4(0u,0u,0u,0u);
      const bool rok = (row0 + ar) < M;
      const long long base = (long long)(row0 + ar) * lda + k0 + akc;
      if (rok && (k0 + akc) < K)     q0 = *reinterpret_cast<const uint4*>(Ab + base);
      if (rok && (k0 + akc + 8) < K) q1 = *reinterpret_cast<const uint4*>(Ab + base + 8);
      *reinterpret_cast<uint4*>(&lsA[ar * 40 + akc])     = q0;
      *reinterpret_cast<uint4*>(&lsA[ar * 40 + akc + 8]) = q1;
    }

    // ---- stage B tile: two consecutive k rows -> packed b32 stores [n][k] ----
    {
      float4 r0 = make_float4(0.f,0.f,0.f,0.f), r1 = r0;
      if ((k0 + bkp) < K) {   // K is even for all GEMMs here -> k+1 valid too
        const float* s0 = Bb + (long long)(k0 + bkp) * ldb + col0 + bng;
        r0 = *reinterpret_cast<const float4*>(s0);
        r1 = *reinterpret_cast<const float4*>(s0 + ldb);
      }
      const unsigned p0 = pack2bf(r0.x, r1.x);
      const unsigned p1 = pack2bf(r0.y, r1.y);
      const unsigned p2 = pack2bf(r0.z, r1.z);
      const unsigned p3 = pack2bf(r0.w, r1.w);
      *reinterpret_cast<unsigned*>(&lsB[(bng + 0) * 40 + bkp]) = p0;
      *reinterpret_cast<unsigned*>(&lsB[(bng + 1) * 40 + bkp]) = p1;
      *reinterpret_cast<unsigned*>(&lsB[(bng + 2) * 40 + bkp]) = p2;
      *reinterpret_cast<unsigned*>(&lsB[(bng + 3) * 40 + bkp]) = p3;
    }

    __syncthreads();

    // ---- fragments per documented 16-bit WMMA layouts ----
    union Frag { bf16x16 v; uint4 q[2]; } fa[2], fb[2];
    #pragma unroll
    for (int i = 0; i < 2; ++i) {
      const unsigned short* ap = &lsA[(wm * 32 + i * 16 + lm) * 40 + lh * 8];
      fa[i].q[0] = *reinterpret_cast<const uint4*>(ap);        // K = lh*8 .. +7
      fa[i].q[1] = *reinterpret_cast<const uint4*>(ap + 16);   // K = 16+lh*8 .. +7
    }
    #pragma unroll
    for (int j = 0; j < 2; ++j) {
      const unsigned short* bp = &lsB[(wn * 32 + j * 16 + lm) * 40 + lh * 8];
      fb[j].q[0] = *reinterpret_cast<const uint4*>(bp);        // K = lh*8 .. +7
      fb[j].q[1] = *reinterpret_cast<const uint4*>(bp + 16);   // K = 16+lh*8 .. +7
    }

    #pragma unroll
    for (int i = 0; i < 2; ++i)
      #pragma unroll
      for (int j = 0; j < 2; ++j)
        acc[i][j] = __builtin_amdgcn_wmma_f32_16x16x32_bf16(
            false, fa[i].v, false, fb[j].v, (short)0, acc[i][j], false, false);

    __syncthreads();
  }

  // ---- epilogue: bias / GELU(exact erf) / residual / f32 + bf16 stores ----
  const long long cb = (long long)bz * sC;
  #pragma unroll
  for (int i = 0; i < 2; ++i) {
    #pragma unroll
    for (int j = 0; j < 2; ++j) {
      const v8f a = acc[i][j];
      const int col = col0 + wn * 32 + j * 16 + lm;
      const float bvv = bias ? bias[col] : 0.0f;
      #pragma unroll
      for (int r = 0; r < 8; ++r) {
        const int row = row0 + wm * 32 + i * 16 + lh * 8 + r;
        if (row < M) {
          float c = a[r] + bvv;
          if (act_gelu) c = 0.5f * c * (1.0f + erff(c * 0.70710678118654752f));
          const long long off = cb + (long long)row * N + col;
          if (resid) c += resid[off];
          if (out32) out32[off] = c;
          if (out16) out16[off] = f2bf(c);
        }
      }
    }
  }
}

// ---------------------------------------------------------------------------
// RBF attention: one block per (b, i) row; softmax over j (196 valid, pad 208).
// ---------------------------------------------------------------------------
__global__ __launch_bounds__(256)
void attn_softmax_kernel(const float* __restrict__ A_noise, const float* __restrict__ b_noise,
                         const float* __restrict__ A_mean,  const float* __restrict__ b_mean,
                         const float* __restrict__ A_std,   const float* __restrict__ b_std,
                         const float* __restrict__ band_width,
                         unsigned short* __restrict__ attn)
{
  const int i   = blockIdx.x;   // 0..207
  const int b   = blockIdx.y;   // 0..31
  const int tid = threadIdx.x;  // 0..255
  __shared__ float red[256];

  const float a00 = A_mean[0] + softplus_f(A_std[0]) * A_noise[b * 4 + 0];
  const float a01 = A_mean[1] + softplus_f(A_std[1]) * A_noise[b * 4 + 1];
  const float a10 = A_mean[2] + softplus_f(A_std[2]) * A_noise[b * 4 + 2];
  const float a11 = A_mean[3] + softplus_f(A_std[3]) * A_noise[b * 4 + 3];
  const float bt0 = b_mean[0] + softplus_f(b_std[0]) * b_noise[b * 2 + 0];
  const float bt1 = b_mean[1] + softplus_f(b_std[1]) * b_noise[b * 2 + 1];
  const float inv_bw = 1.0f / band_width[0];

  const float xi0 = (float)(i % 14);
  const float xi1 = (float)(i / 14);

  const int  j     = tid;
  const bool valid = (i < NPATCH) && (j < NPATCH);
  float dot = -3.0e38f;
  if (valid) {
    const float xj0 = (float)(j % 14);
    const float xj1 = (float)(j / 14);
    const float t0 = xj0 * a00 + xj1 * a10 + bt0;
    const float t1 = xj0 * a01 + xj1 * a11 + bt1;
    const float d0 = t0 - xi0, d1 = t1 - xi1;
    dot = -(d0 * d0 + d1 * d1) * inv_bw;
  }
  red[tid] = dot; __syncthreads();
  for (int o = 128; o > 0; o >>= 1) { if (tid < o) red[tid] = fmaxf(red[tid], red[tid + o]); __syncthreads(); }
  const float m = red[0]; __syncthreads();
  const float e = valid ? expf(dot - m) : 0.0f;
  red[tid] = e; __syncthreads();
  for (int o = 128; o > 0; o >>= 1) { if (tid < o) red[tid] += red[tid + o]; __syncthreads(); }
  const float s = red[0];
  if (j < NPAD) {
    const float v = (valid && s > 0.f) ? (e / s) : 0.0f;
    attn[((long long)b * NPAD + i) * NPAD + j] = f2bf(v);
  }
}

// ---------------------------------------------------------------------------
// LayerNorm: one block per row (768 = 3 x 256). fp32 in -> bf16 out.
// ---------------------------------------------------------------------------
__global__ __launch_bounds__(256)
void ln_kernel(const float* __restrict__ x, const float* __restrict__ g,
               const float* __restrict__ bb, unsigned short* __restrict__ y)
{
  const int row = blockIdx.x;
  const int tid = threadIdx.x;
  const float* xr = x + (long long)row * DIMD;
  __shared__ float red[256];
  const float v0 = xr[tid], v1 = xr[tid + 256], v2 = xr[tid + 512];
  red[tid] = v0 + v1 + v2; __syncthreads();
  for (int o = 128; o > 0; o >>= 1) { if (tid < o) red[tid] += red[tid + o]; __syncthreads(); }
  const float mu = red[0] * (1.0f / 768.0f); __syncthreads();
  const float d0 = v0 - mu, d1 = v1 - mu, d2 = v2 - mu;
  red[tid] = d0 * d0 + d1 * d1 + d2 * d2; __syncthreads();
  for (int o = 128; o > 0; o >>= 1) { if (tid < o) red[tid] += red[tid + o]; __syncthreads(); }
  const float rs = rsqrtf(red[0] * (1.0f / 768.0f) + 1e-5f);
  unsigned short* yr = y + (long long)row * DIMD;
  yr[tid]       = f2bf(d0 * rs * g[tid]       + bb[tid]);
  yr[tid + 256] = f2bf(d1 * rs * g[tid + 256] + bb[tid + 256]);
  yr[tid + 512] = f2bf(d2 * rs * g[tid + 512] + bb[tid + 512]);
}

// ---------------------------------------------------------------------------
// Patchify (b,c,224,224) -> bf16 (b*208, 768); pad rows zero.
// ---------------------------------------------------------------------------
__global__ __launch_bounds__(256)
void patchify_kernel(const float* __restrict__ img, unsigned short* __restrict__ pb)
{
  const long long idx = (long long)blockIdx.x * 256 + threadIdx.x;
  if (idx >= (long long)MTOT * DIMD) return;
  const int pd = (int)(idx % DIMD);
  const long long rq = idx / DIMD;
  const int n = (int)(rq % NPAD);
  const int b = (int)(rq / NPAD);
  float v = 0.0f;
  if (n < NPATCH) {
    const int c  = pd % 3;
    const int t  = pd / 3;
    const int p2 = t % 16;
    const int p1 = t / 16;
    const int h  = n / 14, w = n % 14;
    v = img[(((long long)b * 3 + c) * 224 + (h * 16 + p1)) * 224 + (w * 16 + p2)];
  }
  pb[idx] = f2bf(v);
}

// ---------------------------------------------------------------------------
// Un-patchify (b*208, 768) fp32 -> (b,c,224,224) fp32.
// ---------------------------------------------------------------------------
__global__ __launch_bounds__(256)
void unpatchify_kernel(const float* __restrict__ e, float* __restrict__ out)
{
  const long long idx = (long long)blockIdx.x * 256 + threadIdx.x;
  if (idx >= (long long)B_SZ * 3 * 224 * 224) return;
  const int ww = (int)(idx % 224);
  long long t = idx / 224;
  const int hh = (int)(t % 224); t /= 224;
  const int c  = (int)(t % 3);
  const int b  = (int)(t / 3);
  const int h = hh / 16, p1 = hh % 16, w = ww / 16, p2 = ww % 16;
  const int n  = h * 14 + w;
  const int pd = (p1 * 16 + p2) * 3 + c;
  out[idx] = e[((long long)(b * NPAD + n)) * DIMD + pd];
}

// ---------------------------------------------------------------------------
extern "C" void kernel_launch(void* const* d_in, const int* in_sizes, int n_in,
                              void* d_out, int out_size, void* d_ws, size_t ws_size,
                              hipStream_t stream)
{
  (void)in_sizes; (void)n_in; (void)out_size; (void)ws_size;
  const float* img     = (const float*)d_in[0];
  const float* A_noise = (const float*)d_in[2];
  const float* b_noise = (const float*)d_in[3];
  const float* A_mean  = (const float*)d_in[4];
  const float* b_mean  = (const float*)d_in[5];
  const float* A_std   = (const float*)d_in[6];
  const float* b_std   = (const float*)d_in[7];
  const float* band_w  = (const float*)d_in[8];
  const float* Wp      = (const float*)d_in[9];
  const float* bp      = (const float*)d_in[10];
  const float* ln1_g   = (const float*)d_in[11];
  const float* ln1_b   = (const float*)d_in[12];
  const float* Wqkv    = (const float*)d_in[13];
  const float* Wout    = (const float*)d_in[14];
  const float* bout    = (const float*)d_in[15];
  const float* ln2_g   = (const float*)d_in[16];
  const float* ln2_b   = (const float*)d_in[17];
  const float* Wff1    = (const float*)d_in[18];
  const float* bff1    = (const float*)d_in[19];
  const float* Wff2    = (const float*)d_in[20];
  const float* bff2    = (const float*)d_in[21];
  const float* Wep     = (const float*)d_in[22];
  const float* bep     = (const float*)d_in[23];

  char* p = (char*)d_ws;
  float* xf32          = (float*)p;          p += (size_t)MTOT * DIMD * 4;
  float* vf32          = (float*)p;          p += (size_t)MTOT * DIMD * 4;
  unsigned short* yb   = (unsigned short*)p; p += (size_t)MTOT * DIMD * 2;
  unsigned short* xb   = (unsigned short*)p; p += (size_t)MTOT * DIMD * 2;
  unsigned short* at   = (unsigned short*)p; p += (size_t)B_SZ * NPAD * NPAD * 2;
  unsigned short* hb   = (unsigned short*)p; p += (size_t)MTOT * MLPD * 2;
  unsigned short* ob   = hb;    // alias: lifetimes disjoint within a layer
  float* ef32          = vf32;  // alias: V-buffer free at the end

  const dim3 blk(256);

  patchify_kernel<<<dim3((unsigned)(((long long)MTOT * DIMD + 255) / 256)), blk, 0, stream>>>(img, yb);
  attn_softmax_kernel<<<dim3(NPAD, B_SZ), blk, 0, stream>>>(
      A_noise, b_noise, A_mean, b_mean, A_std, b_std, band_w, at);

  // x = patch @ Wp + bp
  gemm_bf16_wmma<<<dim3(DIMD / 64, MTOT / 128, 1), blk, 0, stream>>>(
      yb, DIMD, 0, Wp, DIMD, 0, bp, nullptr, xf32, nullptr, 0, MTOT, DIMD, DIMD, 0);

  for (int l = 0; l < 6; ++l) {
    // y = LN1(x)
    ln_kernel<<<dim3(MTOT), blk, 0, stream>>>(xf32, ln1_g + l * DIMD, ln1_b + l * DIMD, yb);
    // V = y @ Wqkv[:, 2*INNER:]   (skip dead Q,K)
    gemm_bf16_wmma<<<dim3(DIMD / 64, MTOT / 128, 1), blk, 0, stream>>>(
        yb, DIMD, 0, Wqkv + (size_t)l * DIMD * 3 * DIMD + 2 * DIMD, 3 * DIMD, 0,
        nullptr, nullptr, vf32, nullptr, 0, MTOT, DIMD, DIMD, 0);
    // O = attn @ V   (batched over z = 32)
    gemm_bf16_wmma<<<dim3(DIMD / 64, (NPAD + 127) / 128, B_SZ), blk, 0, stream>>>(
        at, NPAD, (long long)NPAD * NPAD, vf32, DIMD, (long long)NPAD * DIMD,
        nullptr, nullptr, nullptr, ob, (long long)NPAD * DIMD, NPAD, DIMD, NPAD, 0);
    // x = O @ Wout + bout   (no residual, per reference)
    gemm_bf16_wmma<<<dim3(DIMD / 64, MTOT / 128, 1), blk, 0, stream>>>(
        ob, DIMD, 0, Wout + (size_t)l * DIMD * DIMD, DIMD, 0,
        bout + (size_t)l * DIMD, nullptr, xf32, nullptr, 0, MTOT, DIMD, DIMD, 0);
    // h = LN2(x)
    ln_kernel<<<dim3(MTOT), blk, 0, stream>>>(xf32, ln2_g + l * DIMD, ln2_b + l * DIMD, yb);
    // H = gelu(h @ Wff1 + bff1)
    gemm_bf16_wmma<<<dim3(MLPD / 64, MTOT / 128, 1), blk, 0, stream>>>(
        yb, DIMD, 0, Wff1 + (size_t)l * DIMD * MLPD, MLPD, 0,
        bff1 + (size_t)l * MLPD, nullptr, nullptr, hb, 0, MTOT, MLPD, DIMD, 1);
    // x = x + H @ Wff2 + bff2  (dual store: fp32 master + bf16 for next GEMM)
    gemm_bf16_wmma<<<dim3(DIMD / 64, MTOT / 128, 1), blk, 0, stream>>>(
        hb, MLPD, 0, Wff2 + (size_t)l * MLPD * DIMD, DIMD, 0,
        bff2 + (size_t)l * DIMD, xf32, xf32, xb, 0, MTOT, DIMD, MLPD, 0);
  }

  // e = x @ Wep + bep
  gemm_bf16_wmma<<<dim3(DIMD / 64, MTOT / 128, 1), blk, 0, stream>>>(
      xb, DIMD, 0, Wep, DIMD, 0, bep, nullptr, ef32, nullptr, 0, MTOT, DIMD, DIMD, 0);

  unpatchify_kernel<<<dim3((unsigned)(((long long)B_SZ * 3 * 224 * 224 + 255) / 256)), blk, 0, stream>>>(
      ef32, (float*)d_out);
}
